// SABlock_14998025797906
// MI455X (gfx1250) — matrix operand order repaired and checked by reading, not compiled, not run on hardware
//
#include <hip/hip_runtime.h>
#include <hip/hip_bf16.h>
#include <math.h>
#include <stdint.h>

// ---------------------------------------------------------------------------
// CDNA5 (gfx1250) causal self-attention block.
// bf16 WMMA (f32 accum) + async global->LDS double buffering (ASYNCcnt).
// ---------------------------------------------------------------------------

typedef __bf16 bf16_t;
typedef __attribute__((ext_vector_type(16))) __bf16 v16bf;
typedef __attribute__((ext_vector_type(8)))  __bf16 v8bf;
typedef __attribute__((ext_vector_type(4)))  __bf16 v4bf;
typedef __attribute__((ext_vector_type(8)))  float  v8f;

#define WMMA_BF16(a, b, c) \
    __builtin_amdgcn_wmma_f32_16x16x32_bf16(false, (a), false, (b), (short)0, (c), false, false)

static constexpr int Bb = 4, Tt = 2048, Cc = 768, Hh = 12, Dh = 64;
static constexpr float SCALE = 0.125f;  // 64^-0.5

static __device__ inline v16bf cat8(v8bf lo, v8bf hi) {
  return __builtin_shufflevector(lo, hi, 0,1,2,3,4,5,6,7,8,9,10,11,12,13,14,15);
}

// A-fragment (16x32 bf16, ISA 7.12.2): lane holds row m = lane&15;
// halves 0..7 -> k = hi*8 + e ; halves 8..15 -> k = 16 + hi*8 + e.
static __device__ inline v16bf load_afrag(const bf16_t* rowk, int hi) {
  v8bf lo = *(const v8bf*)(rowk + hi * 8);
  v8bf h2 = *(const v8bf*)(rowk + 16 + hi * 8);
  return cat8(lo, h2);
}

// B-fragment (32x16 bf16): lane holds column n = lane&15;
// halves 0..7 -> k = hi*8 + e ; halves 8..15 -> k = 16 + hi*8 + e.
static __device__ inline v16bf load_bfrag(const bf16_t* rowk, int hi) {
  v8bf lo = *(const v8bf*)(rowk + hi * 8);
  v8bf h2 = *(const v8bf*)(rowk + 16 + hi * 8);
  return cat8(lo, h2);
}

// Async global->LDS 16B copy (CDNA5, tracked by ASYNCcnt).
// vdst VGPR = LDS byte address; vaddr pair = 64-bit global address.
static __device__ inline void async_ld_b128(uint32_t lds_addr, const void* gaddr) {
  asm volatile("global_load_async_to_lds_b128 %0, %1, off"
               :: "v"(lds_addr), "v"(gaddr) : "memory");
}
static __device__ inline void wait_async_le4() {
  asm volatile("s_wait_asynccnt 0x4" ::: "memory");
}
static __device__ inline void wait_async_0() {
  asm volatile("s_wait_asynccnt 0x0" ::: "memory");
}
// Compiler-only memory fence: HW keeps same-wave LDS ops in order (DScnt).
static __device__ inline void wave_lds_fence() { asm volatile("" ::: "memory"); }

// ---------------------------------------------------------------------------
// f32 -> bf16 conversion (vector 4-wide)
// ---------------------------------------------------------------------------
__global__ void cvt_f32_bf16(const float* __restrict__ in, bf16_t* __restrict__ out, int n4) {
  int i = blockIdx.x * blockDim.x + threadIdx.x;
  if (i < n4) {
    float4 v = ((const float4*)in)[i];
    v4bf o;
    o[0] = (bf16_t)v.x; o[1] = (bf16_t)v.y; o[2] = (bf16_t)v.z; o[3] = (bf16_t)v.w;
    ((v4bf*)out)[i] = o;
  }
}

// ---------------------------------------------------------------------------
// Tiled bf16 GEMM: C[M,N] = A[M,K] * Bt[N,K]^T  (Bt row-major = torch weight)
// 128x128x32 block tile, 256 threads (8 waves, 4x2), wave tile 32x64.
// Double-buffered LDS filled by async global->LDS copies.
// MODE 0: QKV epilogue (scatter q*SCALE, k, v^T bf16). MODE 1: f32 + bias.
// ---------------------------------------------------------------------------
#define BM 128
#define BN 128
#define BK 32
#define LDS_STR 40  // padded row stride in halves (80B: keeps 16B alignment)

template <int MODE>
__global__ __launch_bounds__(256) void gemm_bf16(
    const bf16_t* __restrict__ A, const bf16_t* __restrict__ Bt,
    int M, int N, int K,
    float* __restrict__ outf, const float* __restrict__ bias,
    bf16_t* __restrict__ qbuf, bf16_t* __restrict__ kbuf, bf16_t* __restrict__ vtbuf) {
  __shared__ __align__(16) bf16_t As[2][BM * LDS_STR];
  __shared__ __align__(16) bf16_t Bs[2][BN * LDS_STR];

  const int tid  = threadIdx.x;
  const int lane = tid & 31, wave = tid >> 5;
  const int hi = lane >> 4, lm = lane & 15;
  const int waveM = wave >> 1;      // 0..3
  const int waveN = wave & 1;       // 0..1
  const int m0 = blockIdx.y * BM, n0 = blockIdx.x * BN;

  const int ldr = tid >> 1;         // 0..127 : row staged by this thread
  const int ldc = (tid & 1) * 16;   // 0 or 16: 16-half chunk

  // Issue one K-tile's async copies (4 x b128 per thread -> ASYNCcnt +4).
  auto issue_tile = [&](int buf, int k0) {
    const bf16_t* ga = &A[(size_t)(m0 + ldr) * K + k0 + ldc];
    const bf16_t* gb = &Bt[(size_t)(n0 + ldr) * K + k0 + ldc];
    uint32_t la = (uint32_t)(uintptr_t)(const void*)&As[buf][ldr * LDS_STR + ldc];
    uint32_t lb = (uint32_t)(uintptr_t)(const void*)&Bs[buf][ldr * LDS_STR + ldc];
    async_ld_b128(la,      ga);
    async_ld_b128(la + 16, ga + 8);
    async_ld_b128(lb,      gb);
    async_ld_b128(lb + 16, gb + 8);
  };

  v8f acc[2][4];
  for (int mt = 0; mt < 2; ++mt)
    for (int nt = 0; nt < 4; ++nt)
      for (int r = 0; r < 8; ++r) acc[mt][nt][r] = 0.0f;

  const int ksteps = K / BK;
  issue_tile(0, 0);

  for (int kt = 0; kt < ksteps; ++kt) {
    const int cur = kt & 1;
    if (kt + 1 < ksteps) {
      issue_tile(cur ^ 1, (kt + 1) * BK);  // prefetch next tile
      wait_async_le4();                    // current tile landed (in-order)
    } else {
      wait_async_0();
    }
    __syncthreads();                       // all waves' copies visible

    v16bf af[2], bf[4];
    for (int mt = 0; mt < 2; ++mt)
      af[mt] = load_afrag(&As[cur][(waveM * 32 + mt * 16 + lm) * LDS_STR], hi);
    for (int nt = 0; nt < 4; ++nt)
      bf[nt] = load_bfrag(&Bs[cur][(waveN * 64 + nt * 16 + lm) * LDS_STR], hi);
    for (int mt = 0; mt < 2; ++mt)
      for (int nt = 0; nt < 4; ++nt)
        acc[mt][nt] = WMMA_BF16(af[mt], bf[nt], acc[mt][nt]);

    __syncthreads();                       // done reading before next overwrite
  }

  // Epilogue. C/D layout: element r of a 16x16 tile is (m = r + 8*hi, n = lm).
  for (int mt = 0; mt < 2; ++mt) {
    for (int nt = 0; nt < 4; ++nt) {
      const int mgb = m0 + waveM * 32 + mt * 16 + 8 * hi;
      const int ng  = n0 + waveN * 64 + nt * 16 + lm;
      if (MODE == 1) {
        const float bv = bias[ng];
        for (int r = 0; r < 8; ++r)
          outf[(size_t)(mgb + r) * N + ng] = acc[mt][nt][r] + bv;
      } else {
        const int which = ng / Cc;          // 0=q 1=k 2=v
        const int rem   = ng - which * Cc;
        const int h     = rem >> 6;         // /64
        const int d     = rem & 63;
        for (int r = 0; r < 8; ++r) {
          const int mg = mgb + r;
          const int b  = mg >> 11;          // /2048
          const int t  = mg & 2047;
          const size_t bh = (size_t)(b * Hh + h);
          const float v = acc[mt][nt][r];
          if (which == 0)      qbuf[(bh * Tt + t) * Dh + d] = (bf16_t)(v * SCALE);
          else if (which == 1) kbuf[(bh * Tt + t) * Dh + d] = (bf16_t)v;
          else                 vtbuf[(bh * Dh + d) * Tt + t] = (bf16_t)v;  // V^T
        }
      }
    }
  }
}

// ---------------------------------------------------------------------------
// Flash attention (causal, online softmax). Grid = (T/128, B*H), 256 threads.
// Each of 8 waves owns 16 query rows with its OWN causal loop bound (no block
// barrier needed: P-staging LDS is per-wave and same-wave LDS ops are in-order
// per the DScnt rules).
// ---------------------------------------------------------------------------
#define P_STR 72  // padded LDS row stride in halves (144B, 16B-aligned chunks)

__global__ __launch_bounds__(256) void attn_flash(
    const bf16_t* __restrict__ q, const bf16_t* __restrict__ k,
    const bf16_t* __restrict__ vt, bf16_t* __restrict__ att) {
  __shared__ __align__(16) bf16_t Pst[8][16 * P_STR];

  const int tid  = threadIdx.x;
  const int lane = tid & 31, wave = tid >> 5;
  const int hi = lane >> 4, lm = lane & 15;
  const int bh = blockIdx.y;                       // b*H + h
  const int qrow0 = blockIdx.x * 128 + wave * 16;  // this wave's query rows
  const int jbmax = (qrow0 + 15) >> 6;             // per-wave causal bound

  // Q A-fragments for this wave's 16 rows, K = 64 -> two 16x32 fragments.
  const bf16_t* qrow = q + ((size_t)bh * Tt + qrow0 + lm) * Dh;
  const v16bf aq0 = load_afrag(qrow, hi);
  const v16bf aq1 = load_afrag(qrow + 32, hi);

  const bf16_t* kb = k + (size_t)bh * Tt * Dh;
  const bf16_t* vb = vt + (size_t)bh * Dh * Tt;

  float rmax[8], rsum[8];
  v8f oacc[4];
  for (int r = 0; r < 8; ++r) { rmax[r] = -INFINITY; rsum[r] = 0.0f; }
  for (int nt = 0; nt < 4; ++nt)
    for (int r = 0; r < 8; ++r) oacc[nt][r] = 0.0f;

  for (int jb = 0; jb <= jbmax; ++jb) {
    const int t0 = jb * 64;

    // ---- S = Q * K^T (Q pre-scaled by D^-1/2) ----
    v8f s[4];
    for (int nt = 0; nt < 4; ++nt) {
      const bf16_t* krow = kb + (size_t)(t0 + nt * 16 + lm) * Dh;
      v8f a; for (int r = 0; r < 8; ++r) a[r] = 0.0f;
      a = WMMA_BF16(aq0, load_bfrag(krow, hi), a);
      a = WMMA_BF16(aq1, load_bfrag(krow + 32, hi), a);
      s[nt] = a;
    }

    // ---- causal mask on the diagonal block ----
    if (jb == jbmax) {
      for (int nt = 0; nt < 4; ++nt) {
        const int tk = t0 + nt * 16 + lm;
        for (int r = 0; r < 8; ++r) {
          const int tq = qrow0 + r + 8 * hi;
          if (tk > tq) s[nt][r] = -INFINITY;
        }
      }
    }

    // ---- online softmax: row max over 64 columns (16-lane half reduce) ----
    float mnew[8];
    for (int r = 0; r < 8; ++r) {
      float m = rmax[r];
      for (int nt = 0; nt < 4; ++nt) m = fmaxf(m, s[nt][r]);
      mnew[r] = m;
    }
    for (int msk = 1; msk < 16; msk <<= 1)
      for (int r = 0; r < 8; ++r)
        mnew[r] = fmaxf(mnew[r], __shfl_xor(mnew[r], msk, 32));

    float fac[8];
    for (int r = 0; r < 8; ++r) {
      fac[r] = __expf(rmax[r] - mnew[r]);
      rmax[r] = mnew[r];
      rsum[r] *= fac[r];
    }
    for (int nt = 0; nt < 4; ++nt)
      for (int r = 0; r < 8; ++r) oacc[nt][r] *= fac[r];

    // ---- P = exp(S - m), stage bf16 to LDS (C-layout -> A-layout bridge) ----
    float psum[8];
    for (int r = 0; r < 8; ++r) psum[r] = 0.0f;
    for (int nt = 0; nt < 4; ++nt) {
      for (int r = 0; r < 8; ++r) {
        const float p = __expf(s[nt][r] - mnew[r]);
        psum[r] += p;
        Pst[wave][(r + 8 * hi) * P_STR + nt * 16 + lm] = (bf16_t)p;
      }
    }
    for (int msk = 1; msk < 16; msk <<= 1)
      for (int r = 0; r < 8; ++r) psum[r] += __shfl_xor(psum[r], msk, 32);
    for (int r = 0; r < 8; ++r) rsum[r] += psum[r];

    wave_lds_fence();  // per-wave region; HW DS ordering handles the rest

    // ---- O += P * V  (V stored transposed: B[k=t][n=d] contiguous in k) ----
    for (int ks = 0; ks < 2; ++ks) {
      const v16bf pa = load_afrag(&Pst[wave][lm * P_STR + ks * 32], hi);
      for (int nt = 0; nt < 4; ++nt) {
        const bf16_t* vrow = vb + (size_t)(nt * 16 + lm) * Tt + t0 + ks * 32;
        oacc[nt] = WMMA_BF16(pa, load_bfrag(vrow, hi), oacc[nt]);
      }
    }
    wave_lds_fence();  // reads done (in program+DS order) before next overwrite
  }

  // ---- normalize and store to att[B,T,C] (head-interleaved layout) ----
  const int b = bh / Hh, h = bh - b * Hh;
  for (int r = 0; r < 8; ++r) {
    const int tq = qrow0 + r + 8 * hi;
    const float inv = 1.0f / rsum[r];
    for (int nt = 0; nt < 4; ++nt) {
      att[(size_t)(b * Tt + tq) * Cc + h * 64 + nt * 16 + lm] =
          (bf16_t)(oacc[nt][r] * inv);
    }
  }
}

// ---------------------------------------------------------------------------
// Host launcher
// ---------------------------------------------------------------------------
extern "C" void kernel_launch(void* const* d_in, const int* in_sizes, int n_in,
                              void* d_out, int out_size, void* d_ws, size_t ws_size,
                              hipStream_t stream) {
  const float* x     = (const float*)d_in[0];  // [B,T,C]
  const float* w_qkv = (const float*)d_in[1];  // [3C,C]
  const float* w_out = (const float*)d_in[2];  // [C,C]
  const float* b_out = (const float*)d_in[3];  // [C]
  float* out = (float*)d_out;

  const size_t MT = (size_t)Bb * Tt;           // 8192
  char* ws = (char*)d_ws;
  bf16_t* xb   = (bf16_t*)ws;  ws += MT * Cc * sizeof(bf16_t);
  bf16_t* wqb  = (bf16_t*)ws;  ws += (size_t)3 * Cc * Cc * sizeof(bf16_t);
  bf16_t* wob  = (bf16_t*)ws;  ws += (size_t)Cc * Cc * sizeof(bf16_t);
  bf16_t* qb_  = (bf16_t*)ws;  ws += (size_t)Bb * Hh * Tt * Dh * sizeof(bf16_t);
  bf16_t* kb_  = (bf16_t*)ws;  ws += (size_t)Bb * Hh * Tt * Dh * sizeof(bf16_t);
  bf16_t* vtb  = (bf16_t*)ws;  ws += (size_t)Bb * Hh * Dh * Tt * sizeof(bf16_t);
  bf16_t* attb = (bf16_t*)ws;  ws += MT * Cc * sizeof(bf16_t);

  auto cvt = [&](const float* src, bf16_t* dst, size_t n) {
    int n4 = (int)(n / 4);
    cvt_f32_bf16<<<(n4 + 255) / 256, 256, 0, stream>>>(src, dst, n4);
  };
  cvt(x,     xb,  MT * Cc);
  cvt(w_qkv, wqb, (size_t)3 * Cc * Cc);
  cvt(w_out, wob, (size_t)Cc * Cc);

  // QKV projection: M=8192, N=2304, K=768
  gemm_bf16<0><<<dim3(3 * Cc / BN, MT / BM), 256, 0, stream>>>(
      xb, wqb, (int)MT, 3 * Cc, Cc, nullptr, nullptr, qb_, kb_, vtb);

  // Fused causal flash attention: grid (T/128, B*H)
  attn_flash<<<dim3(Tt / 128, Bb * Hh), 256, 0, stream>>>(qb_, kb_, vtb, attb);

  // Output projection + bias: M=8192, N=768, K=768
  gemm_bf16<1><<<dim3(Cc / BN, MT / BM), 256, 0, stream>>>(
      attb, wob, (int)MT, Cc, Cc, out, b_out, nullptr, nullptr, nullptr);
}
